// S5SSM_39350490366359
// MI455X (gfx1250) — matrix operand order repaired and checked by reading, not compiled, not run on hardware
//
#include <hip/hip_runtime.h>
#include <stdint.h>

// ---------------------------------------------------------------------------
// S5 SSM forward: L=32768, H=512, P=256
//   1) setup: Lambda_bar, Lambda_bar^128, gamma=(Lambda_bar-1)/Lambda
//   2) W1 = [Re(gamma*B_tilde) ; Im(gamma*B_tilde)]  (512x512 bf16, row=n, col=k)
//      W2 = [2*C_re | -2*C_im]                       (512x512 bf16, row=h, col=k)
//   3) GEMM1: Bu(L,512) = u(L,512) @ W1^T            (bf16 WMMA, f32 accum)
//   4) chunked linear scan x_t = lb*x_{t-1} + Bu_t   (in-place, C=256 chunks of T=128)
//   5) GEMM2: out = xs(L,512) @ W2^T + D*u           (bf16 WMMA, f32 accum)
// ---------------------------------------------------------------------------

#define L_SEQ 32768
#define H_DIM 512
#define P_DIM 256
#define KD    512      // shared K for both GEMMs (== H == 2P)
#define ND    512      // shared N for both GEMMs
#define CHUNK 128      // scan chunk length
#define NCHUNK (L_SEQ / CHUNK)

typedef __attribute__((ext_vector_type(8)))  float          v8f;
typedef __attribute__((ext_vector_type(16))) __bf16         v16bf;
typedef __attribute__((ext_vector_type(8)))  unsigned short u16x8;
typedef __attribute__((ext_vector_type(16))) unsigned short u16x16;

__device__ __forceinline__ unsigned bf16rn(float x) {
    unsigned u = __builtin_bit_cast(unsigned, x);
    unsigned r = u + 0x7fffu + ((u >> 16) & 1u);   // round-to-nearest-even
    return r >> 16;
}

// ------------------------------ setup --------------------------------------
// consts layout (floats): [0]=lb_re [256]=lb_im [512]=lbT_re [768]=lbT_im
//                         [1024]=g_re [1280]=g_im
__global__ void s5_setup(const float* __restrict__ Lre,
                         const float* __restrict__ Lim,
                         const float* __restrict__ logstep,
                         float* __restrict__ consts) {
    int p = threadIdx.x;
    if (p >= P_DIM) return;
    float a  = fminf(Lre[p], -1e-4f);
    float b  = Lim[p];
    float dt = __expf(logstep[p]);
    float ea = __expf(a * dt);
    float lre = ea * __cosf(b * dt);
    float lim = ea * __sinf(b * dt);
    consts[p]           = lre;
    consts[P_DIM + p]   = lim;
    // lb^CHUNK via 7 squarings (CHUNK = 128 = 2^7)
    float zr = lre, zi = lim;
    #pragma unroll
    for (int i = 0; i < 7; ++i) {
        float nr = zr * zr - zi * zi;
        float ni = 2.f * zr * zi;
        zr = nr; zi = ni;
    }
    consts[2 * P_DIM + p] = zr;
    consts[3 * P_DIM + p] = zi;
    // gamma = (lb - 1) / Lambda   (complex division)
    float nr = lre - 1.0f, ni = lim;
    float den = a * a + b * b;
    consts[4 * P_DIM + p] = (nr * a + ni * b) / den;
    consts[5 * P_DIM + p] = (ni * a - nr * b) / den;
}

// --------------------------- weight prep -----------------------------------
__global__ void s5_prep_w1(const float* __restrict__ Bre,
                           const float* __restrict__ Bim,
                           const float* __restrict__ consts,
                           unsigned short* __restrict__ W1) {
    int idx = blockIdx.x * blockDim.x + threadIdx.x;   // P*H threads
    if (idx >= P_DIM * H_DIM) return;
    int p = idx >> 9;          // /512
    int h = idx & 511;
    float gre = consts[4 * P_DIM + p];
    float gim = consts[5 * P_DIM + p];
    float br  = Bre[idx];
    float bi  = Bim[idx];
    W1[(size_t)p * KD + h]             = (unsigned short)bf16rn(gre * br - gim * bi);
    W1[(size_t)(p + P_DIM) * KD + h]   = (unsigned short)bf16rn(gre * bi + gim * br);
}

__global__ void s5_prep_w2(const float* __restrict__ Cre,
                           const float* __restrict__ Cim,
                           unsigned short* __restrict__ W2) {
    int idx = blockIdx.x * blockDim.x + threadIdx.x;   // H*512 threads
    if (idx >= H_DIM * ND) return;
    int h = idx >> 9;
    int k = idx & 511;
    float v = (k < P_DIM) ? (2.0f * Cre[h * P_DIM + k])
                          : (-2.0f * Cim[h * P_DIM + (k - P_DIM)]);
    W2[(size_t)h * KD + k] = (unsigned short)bf16rn(v);
}

// ------------------------------- GEMM --------------------------------------
// Out(M,512) = A(M,512) @ W^T   (W stored row-major as (N=512, K=512) bf16)
// Block: 256 thr (8 waves); tile 128x128; wave tile 32x64; K staged by 32,
// double-buffered LDS; v_wmma_f32_16x16x32_bf16.
#define BM 128
#define BN 128
#define BK 32
#define LDSPITCH 40   // 32 + 8 bf16 pad; 80B row stride keeps 16B alignment

__global__ __launch_bounds__(256)
void s5_gemm_bf16(const float* __restrict__ A,
                  const unsigned short* __restrict__ W,
                  float* __restrict__ Out,
                  const float* __restrict__ Dvec,   // nullptr for GEMM1
                  const float* __restrict__ U,      // nullptr for GEMM1
                  int M) {
    __shared__ unsigned short sA[2][BM][LDSPITCH];
    __shared__ unsigned short sB[2][BN][LDSPITCH];

    const int tid      = threadIdx.x;
    const int blockRow = blockIdx.x * BM;
    const int blockCol = blockIdx.y * BN;
    const int wid      = tid >> 5;
    const int lane     = tid & 31;
    const int lrow     = lane & 15;
    const bool hihalf  = lane >= 16;
    const int waveM    = (wid & 3) * 32;   // 4 waves along M
    const int waveN    = (wid >> 2) * 64;  // 2 waves along N

    const int nStages = KD / BK;           // 16

    // ---- stage loader: global -> LDS (f32 A converted to bf16) ----
    auto load_stage = [&](int st, int buf) {
        const int k0 = st * BK;
        // A tile: 128 x 32 f32 -> bf16 ; 4 float4 per thread
        #pragma unroll
        for (int i = 0; i < 4; ++i) {
            int id  = tid + i * 256;
            int row = id >> 3;             // 8 float4 per row
            int c4  = id & 7;
            const float* gp = A + (size_t)(blockRow + row) * KD + k0 + c4 * 4;
            if (i == 0 && st + 2 < nStages)
                __builtin_prefetch(gp + 2 * BK, 0, 1);   // global_prefetch_b8
            float4 v = *(const float4*)gp;
            unsigned u0 = bf16rn(v.x) | (bf16rn(v.y) << 16);
            unsigned u1 = bf16rn(v.z) | (bf16rn(v.w) << 16);
            *(unsigned*)&sA[buf][row][c4 * 4]     = u0;
            *(unsigned*)&sA[buf][row][c4 * 4 + 2] = u1;
        }
        // B tile: 128 x 32 bf16 ; 16 bf16 (two 16B chunks) per thread
        {
            int n    = tid >> 1;
            int half = tid & 1;
            const u16x8* src =
                (const u16x8*)(W + (size_t)(blockCol + n) * KD + k0 + half * 16);
            u16x8 w0 = src[0];
            u16x8 w1 = src[1];
            *(u16x8*)&sB[buf][n][half * 16]     = w0;
            *(u16x8*)&sB[buf][n][half * 16 + 8] = w1;
        }
    };

    v8f acc[2][4] = {};

    load_stage(0, 0);
    __syncthreads();

    for (int st = 0; st < nStages; ++st) {
        const int buf = st & 1;
        if (st + 1 < nStages) load_stage(st + 1, buf ^ 1);

        // --- A fragments: 16x32 bf16; lanes 0-15 hold K{0-7,16-23} ---
        v16bf afrag[2];
        #pragma unroll
        for (int mi = 0; mi < 2; ++mi) {
            int r  = waveM + mi * 16 + lrow;
            int e0 = hihalf ? 8 : 0;
            u16x8 lo = *(const u16x8*)&sA[buf][r][e0];
            u16x8 hi = *(const u16x8*)&sA[buf][r][e0 + 16];
            u16x16 cat = __builtin_shufflevector(lo, hi,
                0, 1, 2, 3, 4, 5, 6, 7, 8, 9, 10, 11, 12, 13, 14, 15);
            afrag[mi] = __builtin_bit_cast(v16bf, cat);
        }
        // --- B fragments: 32x16 bf16; lane n holds col n, K{0-7,16-23} / K{8-15,24-31} ---
        v16bf bfrag[4];
        #pragma unroll
        for (int ni = 0; ni < 4; ++ni) {
            int c  = waveN + ni * 16 + lrow;
            int e0 = hihalf ? 8 : 0;
            u16x8 lo = *(const u16x8*)&sB[buf][c][e0];
            u16x8 hi = *(const u16x8*)&sB[buf][c][e0 + 16];
            u16x16 cat = __builtin_shufflevector(lo, hi,
                0, 1, 2, 3, 4, 5, 6, 7, 8, 9, 10, 11, 12, 13, 14, 15);
            bfrag[ni] = __builtin_bit_cast(v16bf, cat);
        }
        // --- 8 WMMAs ---
        #pragma unroll
        for (int mi = 0; mi < 2; ++mi)
            #pragma unroll
            for (int ni = 0; ni < 4; ++ni)
                acc[mi][ni] = __builtin_amdgcn_wmma_f32_16x16x32_bf16(
                    false, afrag[mi], false, bfrag[ni],
                    (short)0, acc[mi][ni], false, false);

        __syncthreads();
    }

    // ---- epilogue: C/D layout -> VGPR r: rows r / r+8, cols = lane&15 ----
    #pragma unroll
    for (int mi = 0; mi < 2; ++mi) {
        #pragma unroll
        for (int ni = 0; ni < 4; ++ni) {
            #pragma unroll
            for (int r = 0; r < 8; ++r) {
                int row = blockRow + waveM + mi * 16 + r + (hihalf ? 8 : 0);
                int col = blockCol + waveN + ni * 16 + lrow;
                float v = acc[mi][ni][r];
                if (Dvec)
                    v += Dvec[col] * U[(size_t)row * H_DIM + col];
                Out[(size_t)row * ND + col] = v;
            }
        }
    }
}

// ------------------------------- scan --------------------------------------
// Phase 1: per-chunk local reduce with x0=0 -> carry[c]
__global__ void s5_scan_reduce(const float* __restrict__ Bu,
                               const float* __restrict__ consts,
                               float* __restrict__ carry) {
    int p = threadIdx.x;        // channel
    int c = blockIdx.x;         // chunk
    float lre = consts[p], lim = consts[P_DIM + p];
    float xr = 0.f, xi = 0.f;
    const float* row = Bu + (size_t)c * CHUNK * ND;
    for (int l = 0; l < CHUNK; ++l) {
        float br = row[(size_t)l * ND + p];
        float bi = row[(size_t)l * ND + P_DIM + p];
        float nr = lre * xr - lim * xi + br;
        float ni = lre * xi + lim * xr + bi;
        xr = nr; xi = ni;
    }
    carry[(size_t)c * ND + p]         = xr;
    carry[(size_t)c * ND + P_DIM + p] = xi;
}

// Phase 2: serial scan over chunk carries with lb^CHUNK; store EXCLUSIVE prefix
__global__ void s5_scan_carry(float* __restrict__ carry,
                              const float* __restrict__ consts) {
    int p = threadIdx.x;
    float lre = consts[2 * P_DIM + p], lim = consts[3 * P_DIM + p];
    float xr = 0.f, xi = 0.f;
    for (int c = 0; c < NCHUNK; ++c) {
        float or_ = carry[(size_t)c * ND + p];
        float oi  = carry[(size_t)c * ND + P_DIM + p];
        carry[(size_t)c * ND + p]         = xr;   // state entering chunk c
        carry[(size_t)c * ND + P_DIM + p] = xi;
        float nr = lre * xr - lim * xi + or_;
        float ni = lre * xi + lim * xr + oi;
        xr = nr; xi = ni;
    }
}

// Phase 3: replay with carried-in state; overwrite Bu with xs (in place)
__global__ void s5_scan_apply(float* __restrict__ Bu,
                              const float* __restrict__ consts,
                              const float* __restrict__ carry) {
    int p = threadIdx.x;
    int c = blockIdx.x;
    float lre = consts[p], lim = consts[P_DIM + p];
    float xr = carry[(size_t)c * ND + p];
    float xi = carry[(size_t)c * ND + P_DIM + p];
    float* row = Bu + (size_t)c * CHUNK * ND;
    for (int l = 0; l < CHUNK; ++l) {
        float br = row[(size_t)l * ND + p];
        float bi = row[(size_t)l * ND + P_DIM + p];
        float nr = lre * xr - lim * xi + br;
        float ni = lre * xi + lim * xr + bi;
        xr = nr; xi = ni;
        row[(size_t)l * ND + p]         = xr;
        row[(size_t)l * ND + P_DIM + p] = xi;
    }
}

// ------------------------------ launch -------------------------------------
extern "C" void kernel_launch(void* const* d_in, const int* in_sizes, int n_in,
                              void* d_out, int out_size, void* d_ws, size_t ws_size,
                              hipStream_t stream) {
    const float* u    = (const float*)d_in[0];   // (L,H)
    const float* Lre  = (const float*)d_in[1];   // (P,)
    const float* Lim  = (const float*)d_in[2];   // (P,)
    const float* Bre  = (const float*)d_in[3];   // (P,H)
    const float* Bim  = (const float*)d_in[4];   // (P,H)
    const float* Cre  = (const float*)d_in[5];   // (H,P)
    const float* Cim  = (const float*)d_in[6];   // (H,P)
    const float* Dv   = (const float*)d_in[7];   // (H,)
    const float* lstp = (const float*)d_in[8];   // (P,)
    float* out = (float*)d_out;

    char* ws = (char*)d_ws;
    float*          consts = (float*)(ws + 0);                       // 6KB
    unsigned short* W1     = (unsigned short*)(ws + 8192);           // 512KB
    unsigned short* W2     = (unsigned short*)(ws + 8192 + 524288);  // 512KB
    float*          Bu     = (float*)(ws + 8192 + 2 * 524288);       // 64MB
    float*          carry  = (float*)(ws + 8192 + 2 * 524288 +
                                      (size_t)L_SEQ * ND * 4);       // 512KB

    s5_setup<<<1, 256, 0, stream>>>(Lre, Lim, lstp, consts);
    s5_prep_w1<<<(P_DIM * H_DIM) / 256, 256, 0, stream>>>(Bre, Bim, consts, W1);
    s5_prep_w2<<<(H_DIM * ND) / 256, 256, 0, stream>>>(Cre, Cim, W2);

    dim3 grid(L_SEQ / BM, ND / BN);   // (256, 4)
    s5_gemm_bf16<<<grid, 256, 0, stream>>>(u, W1, Bu, nullptr, nullptr, L_SEQ);

    s5_scan_reduce<<<NCHUNK, 256, 0, stream>>>(Bu, consts, carry);
    s5_scan_carry<<<1, 256, 0, stream>>>(carry, consts);
    s5_scan_apply<<<NCHUNK, 256, 0, stream>>>(Bu, consts, carry);

    s5_gemm_bf16<<<grid, 256, 0, stream>>>(Bu, W2, out, Dv, u, L_SEQ);
}
